// AnomalyGNN_46514495816148
// MI455X (gfx1250) — matrix operand order repaired and checked by reading, not compiled, not run on hardware
//
#include <hip/hip_runtime.h>
#include <hip/hip_bf16.h>

// ---------------------------------------------------------------------------
// AnomalyGNN for MI455X (gfx1250, wave32, WMMA).
// Fused cosine-sim GEMM (column-split x4 for occupancy) + register-resident
// top-K graph build, GCN layers via WMMA f16->f32.
// ---------------------------------------------------------------------------

typedef __attribute__((ext_vector_type(16))) _Float16 v16h;
typedef __attribute__((ext_vector_type(8)))  _Float16 v8h;
typedef __attribute__((ext_vector_type(8)))  float    v8f;

#define NNODES 8192
#define DIM    64
#define HID    128
#define ODIM   64
#define KNN    16
#define KLIST  (KNN + 1)               // 17
#define SPLIT  4                       // column splits of the sim GEMM
#define NKEDGE (NNODES * KNN)          // 131072
#define NEDGE  (NKEDGE + NNODES)       // 139264
#define OUT_N  (NNODES * DIM)          // 524288
#define THRESH 0.5f
#define LN_EPS 1e-5f

static __device__ __forceinline__ v8f wmma16(v16h a, v16h b, v8f c) {
  // 8 args: (neg_a, A, neg_b, B, c_mod, C, reuse_a, reuse_b)
  return __builtin_amdgcn_wmma_f32_16x16x32_f16(
      false, a, false, b, (short)0, c, false, false);
}

// Register-resident sorted (descending) top-17 insert: gated bubble pass.
// kv/ki must be local arrays accessed only with compile-time indices.
#define TOPK_INSERT(kv, ki, vv, cc)                                          \
  do {                                                                       \
    if ((vv) > kv[KNN]) {                                                    \
      kv[KNN] = (vv); ki[KNN] = (cc);                                        \
      _Pragma("unroll")                                                      \
      for (int _p = KNN; _p > 0; --_p) {                                     \
        if (kv[_p] > kv[_p - 1]) {                                           \
          float _tf = kv[_p - 1]; kv[_p - 1] = kv[_p]; kv[_p] = _tf;         \
          int   _ti = ki[_p - 1]; ki[_p - 1] = ki[_p]; ki[_p] = _ti;         \
        }                                                                    \
      }                                                                      \
    }                                                                        \
  } while (0)

// --------------------------- row normalize (f32 -> f16) --------------------
__global__ __launch_bounds__(64)
void k_rownorm(const float* __restrict__ x, _Float16* __restrict__ xn) {
  __shared__ float sh[64];
  int row = blockIdx.x, c = threadIdx.x;
  float v = x[row * DIM + c];
  sh[c] = v * v;
  __syncthreads();
  #pragma unroll
  for (int s = 32; s > 0; s >>= 1) {
    if (c < s) sh[c] += sh[c + s];
    __syncthreads();
  }
  float inv = rsqrtf(sh[0] + 1e-16f);
  xn[row * DIM + c] = (_Float16)(v * inv);
}

// ------------------- fused sim GEMM + partial top-17 -----------------------
// grid = (NNODES/16, SPLIT); one wave per block. Wave owns a 16-row strip and
// scans NNODES/16/SPLIT column tiles. All 32 lanes scan: 2 lanes per row, 8
// candidate columns each; the two per-row partials merge via __shfl, then the
// per-split top-17 list spills to workspace for the merge kernel.
__global__ __launch_bounds__(32)
void k_sim_topk(const _Float16* __restrict__ xn,
                float* __restrict__ pvals, int* __restrict__ pidx) {
  __shared__ float smc[256];

  const int lane = threadIdx.x;        // 0..31
  const int half = lane >> 4;
  const int q    = lane & 15;
  const int row0 = blockIdx.x * 16;
  const int spl  = blockIdx.y;

  // A strip 16x64 f16, WMMA A layout (two k-steps of 32).
  const _Float16* arow = xn + (size_t)(row0 + q) * DIM;
  v8h a0lo = *(const v8h*)(arow + half * 16);
  v8h a0hi = *(const v8h*)(arow + 8 + half * 16);
  v8h a1lo = *(const v8h*)(arow + 32 + half * 16);
  v8h a1hi = *(const v8h*)(arow + 40 + half * 16);
  v16h a0, a1;
  #pragma unroll
  for (int e = 0; e < 8; ++e) {
    a0[e] = a0lo[e]; a0[8 + e] = a0hi[e];
    a1[e] = a1lo[e]; a1[8 + e] = a1hi[e];
  }

  // Per-lane partial top-17 (row = q, column group = half) in registers.
  float kv[KLIST];
  int   ki[KLIST];
  #pragma unroll
  for (int j = 0; j < KLIST; ++j) { kv[j] = -1e30f; ki[j] = 0; }

  const int TPS  = NNODES / 16 / SPLIT;   // 128 tiles per split
  const int tbeg = spl * TPS, tend = tbeg + TPS;
  for (int t = tbeg; t < tend; ++t) {
    const int col0 = t * 16;
    if (t + 1 < tend)  // prefetch next B tile -> global_prefetch_b8
      __builtin_prefetch(xn + (size_t)(col0 + 16 + q) * DIM + half * 32, 0, 1);

    // B = X^T : column (col0+q) of B is row (col0+q) of xn -> contiguous 32B.
    const _Float16* bptr = xn + (size_t)(col0 + q) * DIM;
    v16h b0 = *(const v16h*)(bptr + half * 16);        // K =  0..31
    v16h b1 = *(const v16h*)(bptr + 32 + half * 16);   // K = 32..63

    v8f c = {};
    c = wmma16(a0, b0, c);
    c = wmma16(a1, b1, c);

    // C layout: lane holds col q, rows r + 8*half.  Park tile in LDS.
    #pragma unroll
    for (int r = 0; r < 8; ++r)
      smc[(r + 8 * half) * 16 + q] = c[r];
    __syncthreads();  // single-wave workgroup: barrier is S_NOP + DS fence

    // Each lane scans 8 candidates of its row (row=q, cols half*8..half*8+7).
    const float4* smrow = (const float4*)&smc[q * 16 + half * 8];
    float4 ra = smrow[0];
    float4 rb = smrow[1];
    const int cb = col0 + half * 8;
    TOPK_INSERT(kv, ki, ra.x, cb + 0);
    TOPK_INSERT(kv, ki, ra.y, cb + 1);
    TOPK_INSERT(kv, ki, ra.z, cb + 2);
    TOPK_INSERT(kv, ki, ra.w, cb + 3);
    TOPK_INSERT(kv, ki, rb.x, cb + 4);
    TOPK_INSERT(kv, ki, rb.y, cb + 5);
    TOPK_INSERT(kv, ki, rb.z, cb + 6);
    TOPK_INSERT(kv, ki, rb.w, cb + 7);
    __syncthreads();
  }

  // Merge: lane m (<16) pulls lane m+16's partial list and folds it in.
  #pragma unroll
  for (int p = 0; p < KLIST; ++p) {
    float pv = __shfl(kv[p], q | 16, 32);
    int   pi = __shfl(ki[p], q | 16, 32);
    if (lane < 16) TOPK_INSERT(kv, ki, pv, pi);
  }

  // Spill this split's top-17 for the row.
  if (lane < 16) {
    const int row = row0 + lane;
    float* pv = pvals + ((size_t)row * SPLIT + spl) * KLIST;
    int*   pi = pidx  + ((size_t)row * SPLIT + spl) * KLIST;
    #pragma unroll
    for (int j = 0; j < KLIST; ++j) { pv[j] = kv[j]; pi[j] = ki[j]; }
  }
}

// ---------------- merge per-split partials, emit edges ---------------------
__global__ __launch_bounds__(64)
void k_topk_merge(const float* __restrict__ pvals, const int* __restrict__ pidx,
                  int* __restrict__ edge_dst, float* __restrict__ edge_w,
                  int* __restrict__ o_src, int* __restrict__ o_dst,
                  float* __restrict__ o_ew) {
  const int row = blockIdx.x * blockDim.x + threadIdx.x;
  if (row >= NNODES) return;

  float kv[KLIST];
  int   ki[KLIST];
  #pragma unroll
  for (int j = 0; j < KLIST; ++j) { kv[j] = -1e30f; ki[j] = 0; }

  const float* pv = pvals + (size_t)row * SPLIT * KLIST;
  const int*   pi = pidx  + (size_t)row * SPLIT * KLIST;
  for (int j = 0; j < SPLIT * KLIST; ++j) {
    float v = pv[j];
    int   d = pi[j];
    TOPK_INSERT(kv, ki, v, d);
  }

  // Emit K edges; entry 0 (self, sim~1) dropped like vals[:,1:].
  #pragma unroll
  for (int j = 1; j <= KNN; ++j) {
    const int   e = row * KNN + (j - 1);
    const float v = kv[j];
    const float w = (v > THRESH) ? v : 0.0f;
    const int   d = ki[j];
    edge_dst[e] = d;  edge_w[e] = w;
    o_src[e] = row;   o_dst[e] = d;   o_ew[e] = w;
  }
}

// ----------------------- self loops + degree init --------------------------
__global__ void k_node_init(float* __restrict__ deg,
                            int* o_src, int* o_dst, float* o_ew) {
  int n = blockIdx.x * blockDim.x + threadIdx.x;
  if (n >= NNODES) return;
  deg[n] = 2.0f;  // constructor self loop + gcn_norm self loop
  o_src[NKEDGE + n] = n;
  o_dst[NKEDGE + n] = n;
  o_ew [NKEDGE + n] = 1.0f;
}

__global__ void k_deg_edges(const int* __restrict__ edge_dst,
                            const float* __restrict__ edge_w,
                            float* __restrict__ deg) {
  int e = blockIdx.x * blockDim.x + threadIdx.x;
  if (e >= NKEDGE) return;
  float w = edge_w[e];
  if (w != 0.0f) atomicAdd(&deg[edge_dst[e]], w);
}

__global__ void k_dinv(const float* __restrict__ deg, float* __restrict__ dinv) {
  int n = blockIdx.x * blockDim.x + threadIdx.x;
  if (n < NNODES) dinv[n] = rsqrtf(deg[n]);
}

// --------------------------- dtype conversion ------------------------------
__global__ void k_f32_to_f16(const float* __restrict__ a,
                             _Float16* __restrict__ b, int n) {
  int i = blockIdx.x * blockDim.x + threadIdx.x;
  if (i < n) b[i] = (_Float16)a[i];
}

// --------------- generic WMMA GEMM: C[MxNcol] = A[MxKd] * B[KdxNcol] -------
__global__ __launch_bounds__(32)
void k_gemm_f16(const _Float16* __restrict__ A, const _Float16* __restrict__ B,
                const float* __restrict__ bias, float* __restrict__ C,
                int Kd, int Ncol) {
  const int lane = threadIdx.x;
  const int half = lane >> 4;
  const int q    = lane & 15;
  const int m0   = blockIdx.x * 16;
  const int n0   = blockIdx.y * 16;

  const _Float16* arow = A + (size_t)(m0 + q) * Kd;
  v8f c = {};
  for (int k0 = 0; k0 < Kd; k0 += 32) {
    v8h alo = *(const v8h*)(arow + k0 + half * 16);
    v8h ahi = *(const v8h*)(arow + k0 + 8 + half * 16);
    v16h a, b;
    #pragma unroll
    for (int e = 0; e < 8; ++e) { a[e] = alo[e]; a[8 + e] = ahi[e]; }
    const _Float16* bp = B + (size_t)(n0 + q);
    #pragma unroll
    for (int e = 0; e < 16; ++e)
      b[e] = bp[(size_t)(k0 + half * 16 + e) * Ncol];
    c = wmma16(a, b, c);
  }
  const int col = n0 + q;
  const float bb = bias ? bias[col] : 0.0f;
  #pragma unroll
  for (int r = 0; r < 8; ++r) {
    int row = m0 + r + 8 * half;
    C[(size_t)row * Ncol + col] = c[r] + bb;
  }
}

// --------------------------- GCN aggregation -------------------------------
__global__ void k_agg_init(const float* __restrict__ h,
                           const float* __restrict__ dinv,
                           float* __restrict__ out, int Cols) {
  int i = blockIdx.x * blockDim.x + threadIdx.x;
  if (i >= NNODES * Cols) return;
  int n = i / Cols;
  float di = dinv[n];
  out[i] = 2.0f * di * di * h[i];  // two unit self loops
}

__global__ void k_agg_edges(const float* __restrict__ h,
                            const int* __restrict__ edge_dst,
                            const float* __restrict__ edge_w,
                            const float* __restrict__ dinv,
                            float* __restrict__ out, int Cols) {
  long long i = (long long)blockIdx.x * blockDim.x + threadIdx.x;
  if (i >= (long long)NKEDGE * Cols) return;
  int e = (int)(i / Cols), c = (int)(i % Cols);
  float w = edge_w[e];
  if (w == 0.0f) return;
  int s = e / KNN, d = edge_dst[e];
  float coef = dinv[s] * w * dinv[d];
  atomicAdd(&out[(size_t)d * Cols + c], coef * h[(size_t)s * Cols + c]);
}

// ------------------- bias + relu + layernorm -> f16 ------------------------
__global__ void k_bias_relu_ln(const float* __restrict__ X,
                               const float* __restrict__ bias,
                               const float* __restrict__ g,
                               const float* __restrict__ be,
                               _Float16* __restrict__ Y, int Cols) {
  __shared__ float sh[128];
  int row = blockIdx.x, c = threadIdx.x;
  float t = X[(size_t)row * Cols + c] + bias[c];
  t = fmaxf(t, 0.0f);
  sh[c] = t;
  __syncthreads();
  for (int s = Cols >> 1; s > 0; s >>= 1) {
    if (c < s) sh[c] += sh[c + s];
    __syncthreads();
  }
  float mu = sh[0] / (float)Cols;
  __syncthreads();
  float dv = t - mu;
  sh[c] = dv * dv;
  __syncthreads();
  for (int s = Cols >> 1; s > 0; s >>= 1) {
    if (c < s) sh[c] += sh[c + s];
    __syncthreads();
  }
  float var = sh[0] / (float)Cols;
  float y = dv * rsqrtf(var + LN_EPS) * g[c] + be[c];
  Y[(size_t)row * Cols + c] = (_Float16)y;
}

// ---------------------------------------------------------------------------
extern "C" void kernel_launch(void* const* d_in, const int* in_sizes, int n_in,
                              void* d_out, int out_size, void* d_ws, size_t ws_size,
                              hipStream_t stream) {
  const float* x   = (const float*)d_in[0];
  const float* W1  = (const float*)d_in[1];
  const float* b1  = (const float*)d_in[2];
  const float* g1  = (const float*)d_in[3];
  const float* be1 = (const float*)d_in[4];
  const float* W2  = (const float*)d_in[5];
  const float* b2  = (const float*)d_in[6];
  const float* g2  = (const float*)d_in[7];
  const float* be2 = (const float*)d_in[8];
  const float* Wp  = (const float*)d_in[9];
  const float* bp  = (const float*)d_in[10];

  float* out = (float*)d_out;                      // [0, OUT_N): projection
  int*   o_src = (int*)(out + OUT_N);              // edge_index row 0
  int*   o_dst = o_src + NEDGE;                    // edge_index row 1
  float* o_ew  = (float*)(o_dst + NEDGE);          // edge weights

  char* ws = (char*)d_ws;
  size_t off = 0;
  auto alloc = [&](size_t bytes) { void* p = ws + off; off = (off + bytes + 255) & ~(size_t)255; return p; };

  _Float16* xn    = (_Float16*)alloc((size_t)NNODES * DIM * 2);
  _Float16* w1h   = (_Float16*)alloc((size_t)DIM * HID * 2);
  _Float16* w2h   = (_Float16*)alloc((size_t)HID * ODIM * 2);
  _Float16* wph   = (_Float16*)alloc((size_t)ODIM * DIM * 2);
  float*    h1    = (float*)   alloc((size_t)NNODES * HID * 4);
  float*    ag1   = (float*)   alloc((size_t)NNODES * HID * 4);
  _Float16* ln1   = (_Float16*)alloc((size_t)NNODES * HID * 2);
  float*    h2    = (float*)   alloc((size_t)NNODES * ODIM * 4);
  float*    ag2   = (float*)   alloc((size_t)NNODES * ODIM * 4);
  _Float16* ln2   = (_Float16*)alloc((size_t)NNODES * ODIM * 2);
  float*    deg   = (float*)   alloc((size_t)NNODES * 4);
  float*    dinv  = (float*)   alloc((size_t)NNODES * 4);
  int*      edst  = (int*)     alloc((size_t)NKEDGE * 4);
  float*    ewgt  = (float*)   alloc((size_t)NKEDGE * 4);
  float*    pvals = (float*)   alloc((size_t)NNODES * SPLIT * KLIST * 4);
  int*      pidx  = (int*)     alloc((size_t)NNODES * SPLIT * KLIST * 4);

  // 1) normalize rows -> f16
  k_rownorm<<<NNODES, 64, 0, stream>>>(x, xn);

  // 2) fused similarity GEMM + partial top-K (WMMA-dominant, 2048 waves)
  k_sim_topk<<<dim3(NNODES / 16, SPLIT), 32, 0, stream>>>(xn, pvals, pidx);
  k_topk_merge<<<(NNODES + 63) / 64, 64, 0, stream>>>(
      pvals, pidx, edst, ewgt, o_src, o_dst, o_ew);

  // 3) degrees + symmetric normalization
  k_node_init<<<(NNODES + 255) / 256, 256, 0, stream>>>(deg, o_src, o_dst, o_ew);
  k_deg_edges<<<(NKEDGE + 255) / 256, 256, 0, stream>>>(edst, ewgt, deg);
  k_dinv<<<(NNODES + 255) / 256, 256, 0, stream>>>(deg, dinv);

  // 4) weights -> f16
  k_f32_to_f16<<<(DIM * HID + 255) / 256, 256, 0, stream>>>(W1, w1h, DIM * HID);
  k_f32_to_f16<<<(HID * ODIM + 255) / 256, 256, 0, stream>>>(W2, w2h, HID * ODIM);
  k_f32_to_f16<<<(ODIM * DIM + 255) / 256, 256, 0, stream>>>(Wp, wph, ODIM * DIM);

  // 5) GCN layer 1: h1 = xn @ W1 ; aggregate ; bias+relu+LN -> ln1 (f16)
  k_gemm_f16<<<dim3(NNODES / 16, HID / 16), 32, 0, stream>>>(xn, w1h, nullptr, h1, DIM, HID);
  k_agg_init<<<(NNODES * HID + 255) / 256, 256, 0, stream>>>(h1, dinv, ag1, HID);
  k_agg_edges<<<(int)(((long long)NKEDGE * HID + 255) / 256), 256, 0, stream>>>(
      h1, edst, ewgt, dinv, ag1, HID);
  k_bias_relu_ln<<<NNODES, HID, 0, stream>>>(ag1, b1, g1, be1, ln1, HID);

  // 6) GCN layer 2
  k_gemm_f16<<<dim3(NNODES / 16, ODIM / 16), 32, 0, stream>>>(ln1, w2h, nullptr, h2, HID, ODIM);
  k_agg_init<<<(NNODES * ODIM + 255) / 256, 256, 0, stream>>>(h2, dinv, ag2, ODIM);
  k_agg_edges<<<(int)(((long long)NKEDGE * ODIM + 255) / 256), 256, 0, stream>>>(
      h2, edst, ewgt, dinv, ag2, ODIM);
  k_bias_relu_ln<<<NNODES, ODIM, 0, stream>>>(ag2, b2, g2, be2, ln2, ODIM);

  // 7) projection: out = ln2 @ Wp + bp
  k_gemm_f16<<<dim3(NNODES / 16, DIM / 16), 32, 0, stream>>>(ln2, wph, bp, out, ODIM, DIM);
}
